// DHHGCN_13606456394159
// MI455X (gfx1250) — matrix operand order repaired and checked by reading, not compiled, not run on hardware
//
#include <hip/hip_runtime.h>
#include <hip/hip_bf16.h>

#define NITEMS 100000
#define NUSER  8192
#define NNZ_ADJ 1600000
#define NBATCH 4096

typedef __attribute__((ext_vector_type(16))) _Float16 v16h;
typedef __attribute__((ext_vector_type(8)))  float    v8f;

// ---------------------------------------------------------------- WMMA core
__device__ __forceinline__ v8f wmma16(v16h a, v16h b, v8f c) {
  // D(16x16,f32) = A(16x32,f16) * B(32x16,f16) + C
  return __builtin_amdgcn_wmma_f32_16x16x32_f16(
      /*neg_a=*/false, a, /*neg_b=*/false, b,
      /*c_mod=*/(short)0, c, /*reuse_a=*/false, /*reuse_b=*/false);
}

// A-matrix lane layout (16-bit A 16x32): lanes 0-15 -> K {k0..k0+7, k0+16..k0+23},
// lanes 16-31 -> +8. Row M = m0 + (lane&15). f32 -> f16 in VALU (v_cvt_pk_f16_f32).
__device__ __forceinline__ v16h load_a16(const float* __restrict__ X, int ld,
                                         int m0, int k0, int lane) {
  const float* p = X + (m0 + (lane & 15)) * ld + k0 + ((lane >> 4) << 3);
  v16h a;
#pragma unroll
  for (int j = 0; j < 8; ++j) {
    a[j]     = (_Float16)p[j];
    a[8 + j] = (_Float16)p[16 + j];
  }
  return a;
}

// Pre-packed B operand: per (k-chunk kc, n-tile nt, lane) 16 contiguous halfs.
__device__ __forceinline__ v16h load_b_packed(const _Float16* __restrict__ P,
                                              int kc, int nt, int lane) {
  return *(const v16h*)(P + ((((kc << 2) + nt) << 5) + lane) * 16);
}

// Pack row-major f32 W[K x 64] into WMMA B lane layout, f16.
// gid = kc*128 + nt*32 + lane ; element e -> W[kc*32 + ((lane>>4)<<3) + e + (e>=8 ? 8 : 0)][nt*16 + (lane&15)]
__global__ void pack_b16_kernel(const float* __restrict__ W, _Float16* __restrict__ P,
                                int K) {
  int gid = blockIdx.x * blockDim.x + threadIdx.x;
  int total = (K >> 5) << 7;
  if (gid >= total) return;
  int lane = gid & 31;
  int nt = (gid >> 5) & 3;
  int kc = gid >> 7;
  const float* p = W + ((kc << 5) + ((lane >> 4) << 3)) * 64 + nt * 16 + (lane & 15);
  v16h b;
#pragma unroll
  for (int e = 0; e < 16; ++e) b[e] = (_Float16)p[(e + (e >= 8 ? 8 : 0)) * 64];
  *(v16h*)(P + (size_t)gid * 16) = b;
}

// ------------------------------------------------------------ simple kernels
// m_p[i,:] = vp_val[i] * pri[vp_col[i],:]  (diag rows: exactly 1 nnz per row)
__global__ void gather_scale_kernel(const int* __restrict__ idx,
                                    const float* __restrict__ val,
                                    const float* __restrict__ tab,
                                    float* __restrict__ out, int n) {
  int gid = blockIdx.x * blockDim.x + threadIdx.x;
  if (gid >= n * 16) return;
  int i = gid >> 4, q = gid & 15;
  int col = idx[n + i];
  float v = val[i];
  float4 t = ((const float4*)tab)[col * 16 + q];
  float4 o = {v * t.x, v * t.y, v * t.z, v * t.w};
  ((float4*)out)[i * 16 + q] = o;
}

__global__ void copy4_kernel(const float4* __restrict__ s, float4* __restrict__ d, int n4) {
  int g = blockIdx.x * blockDim.x + threadIdx.x;
  if (g < n4) d[g] = s[g];
}
__global__ void zero4_kernel(float4* __restrict__ d, int n4) {
  int g = blockIdx.x * blockDim.x + threadIdx.x;
  float4 z = {0.f, 0.f, 0.f, 0.f};
  if (g < n4) d[g] = z;
}

// COO spmm: out[row,:] += val * e[col,:]  (unsorted rows -> f32 global atomics)
__global__ void spmm_atomic_kernel(const int* __restrict__ idx,
                                   const float* __restrict__ val,
                                   const float* __restrict__ e,
                                   float* __restrict__ out, int nnz) {
  int gid = blockIdx.x * blockDim.x + threadIdx.x;
  if (gid >= nnz * 16) return;
  int nz = gid >> 4, q = gid & 15;
  int r = idx[nz], c = idx[nnz + nz];
  float v = val[nz];
  float4 ev = ((const float4*)e)[c * 16 + q];
  float* o = out + (size_t)r * 64 + q * 4;
  atomicAdd(o + 0, v * ev.x);
  atomicAdd(o + 1, v * ev.y);
  atomicAdd(o + 2, v * ev.z);
  atomicAdd(o + 3, v * ev.w);
}

// ------------------------------------------------- out = relu(X@W + b) + X
__global__ void __launch_bounds__(256)
lin_res_kernel(const float* __restrict__ X, const _Float16* __restrict__ Wp,
               const float* __restrict__ bias, float* __restrict__ out, int M) {
  int gw = (int)((blockIdx.x * 256u + threadIdx.x) >> 5);  // wave-uniform
  int lane = threadIdx.x & 31;
  if (gw >= (M >> 4)) return;
  int m0 = gw << 4;
  v16h a0 = load_a16(X, 64, m0, 0, lane);
  v16h a1 = load_a16(X, 64, m0, 32, lane);
#pragma unroll
  for (int nt = 0; nt < 4; ++nt) {
    v8f c = {};
    c = wmma16(a0, load_b_packed(Wp, 0, nt, lane), c);
    c = wmma16(a1, load_b_packed(Wp, 1, nt, lane), c);
    int col = nt * 16 + (lane & 15);
    float bb = bias[col];
    int rbase = m0 + ((lane >> 4) << 3);
#pragma unroll
    for (int r = 0; r < 8; ++r) {
      int row = rbase + r;
      float v = c[r] + bb;
      out[row * 64 + col] = fmaxf(v, 0.f) + X[row * 64 + col];
    }
  }
}

// --------------- gate: base=[e|mp|mc]@W1+b1; g1=tanh(base+mp@W2+b2);
//                 g2=tanh(base+mc@W3+b3); item=e+g1*mp+g2*mc+adj; sum+=item
__global__ void __launch_bounds__(256)
gate_kernel(const float* __restrict__ e, const float* __restrict__ mp,
            const float* __restrict__ mc, const float* __restrict__ adj,
            const _Float16* __restrict__ P1, const float* __restrict__ b1,
            const _Float16* __restrict__ P2, const float* __restrict__ b2,
            const _Float16* __restrict__ P3, const float* __restrict__ b3,
            float* __restrict__ item, float* __restrict__ sum, int M) {
  int gw = (int)((blockIdx.x * 256u + threadIdx.x) >> 5);
  int lane = threadIdx.x & 31;
  if (gw >= (M >> 4)) return;
  int m0 = gw << 4;
  v16h ae0 = load_a16(e, 64, m0, 0, lane),  ae1 = load_a16(e, 64, m0, 32, lane);
  v16h ap0 = load_a16(mp, 64, m0, 0, lane), ap1 = load_a16(mp, 64, m0, 32, lane);
  v16h ac0 = load_a16(mc, 64, m0, 0, lane), ac1 = load_a16(mc, 64, m0, 32, lane);
#pragma unroll
  for (int nt = 0; nt < 4; ++nt) {
    v8f base = {};
    base = wmma16(ae0, load_b_packed(P1, 0, nt, lane), base);  // K 0..63   : e
    base = wmma16(ae1, load_b_packed(P1, 1, nt, lane), base);
    base = wmma16(ap0, load_b_packed(P1, 2, nt, lane), base);  // K 64..127 : m_p
    base = wmma16(ap1, load_b_packed(P1, 3, nt, lane), base);
    base = wmma16(ac0, load_b_packed(P1, 4, nt, lane), base);  // K 128..191: m_c
    base = wmma16(ac1, load_b_packed(P1, 5, nt, lane), base);
    v8f t1 = wmma16(ap0, load_b_packed(P2, 0, nt, lane), base);
    t1     = wmma16(ap1, load_b_packed(P2, 1, nt, lane), t1);
    v8f t2 = wmma16(ac0, load_b_packed(P3, 0, nt, lane), base);
    t2     = wmma16(ac1, load_b_packed(P3, 1, nt, lane), t2);
    int col = nt * 16 + (lane & 15);
    float bb1 = b1[col], bb2 = b2[col], bb3 = b3[col];
    int rbase = m0 + ((lane >> 4) << 3);
#pragma unroll
    for (int r = 0; r < 8; ++r) {
      int row = rbase + r;
      float g1 = tanhf(t1[r] + bb1 + bb2);
      float g2 = tanhf(t2[r] + bb1 + bb3);
      size_t o = (size_t)row * 64 + col;
      float it = e[o] + g1 * mp[o] + g2 * mc[o] + adj[o];
      item[o] = it;
      sum[o] += it;
    }
  }
}

// ------------- Mu2 = conv(8192x8192) @ user(8192x64)
// Double-buffered async global->LDS staging of the 256MB A stream + WMMA.
// B (user) is pre-packed f16 in WMMA lane layout (L2-resident, 1MB).
__global__ void __launch_bounds__(128)
conv_gemm_kernel(const float* __restrict__ A, const _Float16* __restrict__ Bp,
                 float* __restrict__ out) {
  __shared__ __attribute__((aligned(16))) float sA[2][64 * 32];  // 2 x 8KB
  const int tid = threadIdx.x;             // 128 threads = 4 waves
  const int wave = tid >> 5, lane = tid & 31;
  const unsigned row0 = blockIdx.x * 64u;

  v8f acc0 = {}, acc1 = {}, acc2 = {}, acc3 = {};

  // A staging: thread -> (row ar, half ah), 16 f32 = 4x b128 per thread / chunk
  const int ar = tid >> 1, ah = tid & 1;
  const unsigned aLds0 = (unsigned)(size_t)(&sA[0][ar * 32 + ah * 16]);
  const unsigned aLds1 = (unsigned)(size_t)(&sA[1][ar * 32 + ah * 16]);
  const unsigned aGof = ((row0 + (unsigned)ar) * 8192u + (unsigned)ah * 16u) * 4u;
  const int NK = 8192 / 32;

  // prologue: stage K-chunk 0 into buffer 0
#pragma unroll
  for (int j = 0; j < 4; ++j)
    asm volatile("global_load_async_to_lds_b128 %0, %1, %2"
                 :: "v"(aLds0 + (unsigned)j * 16u),
                    "v"(aGof + (unsigned)j * 16u), "s"(A) : "memory");

  for (int i = 0; i < NK; ++i) {
    if (i + 1 < NK) {
      unsigned lds = ((i + 1) & 1) ? aLds1 : aLds0;
      unsigned go = aGof + (unsigned)(i + 1) * 128u;  // 32 floats * 4B per chunk
#pragma unroll
      for (int j = 0; j < 4; ++j)
        asm volatile("global_load_async_to_lds_b128 %0, %1, %2"
                     :: "v"(lds + (unsigned)j * 16u),
                        "v"(go + (unsigned)j * 16u), "s"(A) : "memory");
      // in-order completion: <=4 outstanding => chunk i fully landed
      asm volatile("s_wait_asynccnt 4" ::: "memory");
    } else {
      asm volatile("s_wait_asynccnt 0" ::: "memory");
    }
    __syncthreads();                       // all waves' chunk-i data visible
    const float* buf = sA[i & 1];
    v16h a = load_a16(buf, 32, wave * 16, 0, lane);
    acc0 = wmma16(a, load_b_packed(Bp, i, 0, lane), acc0);
    acc1 = wmma16(a, load_b_packed(Bp, i, 1, lane), acc1);
    acc2 = wmma16(a, load_b_packed(Bp, i, 2, lane), acc2);
    acc3 = wmma16(a, load_b_packed(Bp, i, 3, lane), acc3);
    __syncthreads();                       // all reads done before re-staging
  }
  const int col = lane & 15;
  const int rbase = (int)row0 + wave * 16 + ((lane >> 4) << 3);
#pragma unroll
  for (int r = 0; r < 8; ++r) {
    int row = rbase + r;
    out[row * 64 + col + 0]  = acc0[r];
    out[row * 64 + col + 16] = acc1[r];
    out[row * 64 + col + 32] = acc2[r];
    out[row * 64 + col + 48] = acc3[r];
  }
}

// ----------------------- score[b,dom] = dot(MuF[u], Sum[i]) / 3
__global__ void score_kernel(const float* __restrict__ muA, const float* __restrict__ sA,
                             const int* __restrict__ usA, const int* __restrict__ isA,
                             const float* __restrict__ muB, const float* __restrict__ sB,
                             const int* __restrict__ usB, const int* __restrict__ isB,
                             float* __restrict__ out) {
  int b = blockIdx.x * blockDim.x + threadIdx.x;
  if (b >= NBATCH) return;
  int ua = usA[b], ia = isA[b], ub = usB[b], ib = isB[b];
  float accA = 0.f, accB = 0.f;
#pragma unroll 4
  for (int d = 0; d < 64; ++d) {
    accA += muA[ua * 64 + d] * sA[ia * 64 + d];
    accB += muB[ub * 64 + d] * sB[ib * 64 + d];
  }
  out[b * 2 + 0] = accA * (1.f / 3.f);
  out[b * 2 + 1] = accB * (1.f / 3.f);
}

// ------------------------------------------------------------------- driver
extern "C" void kernel_launch(void* const* d_in, const int* in_sizes, int n_in,
                              void* d_out, int out_size, void* d_ws, size_t ws_size,
                              hipStream_t stream) {
  (void)in_sizes; (void)n_in; (void)out_size; (void)ws_size;
  const float* wv1W = (const float*)d_in[26];
  const float* wv1b = (const float*)d_in[27];
  const float* wv2W = (const float*)d_in[28];
  const float* wv2b = (const float*)d_in[29];
  const float* wv3W = (const float*)d_in[30];
  const float* wv3b = (const float*)d_in[31];

  float* ws = (float*)d_ws;
  const size_t SZ = (size_t)NITEMS * 64;   // 6.4M floats
  const size_t MU = (size_t)NUSER * 64;
  const size_t DSTR = 6 * SZ + 2 * MU;

  // packed-f16 weight region lives after the two f32 domain blocks
  _Float16* pk = (_Float16*)(ws + 2 * DSTR);
  _Float16* pw1   = pk;                 // 192x64 -> 6 kchunks * 2048
  _Float16* pw2   = pw1 + 6 * 2048;     // 64x64  -> 2 * 2048
  _Float16* pw3   = pw2 + 2 * 2048;
  _Float16* plin0 = pw3 + 2 * 2048;     // per-domain linW[1]
  _Float16* plin1 = plin0 + 2 * 2048;
  _Float16* pusr0 = plin1 + 2 * 2048;   // per-domain user: 256 kchunks * 2048
  _Float16* pusr1 = pusr0 + 256 * 2048;

  const int T = 256;
  const int gElem  = (NITEMS * 16 + T - 1) / T;
  const int gNNZ   = (NNZ_ADJ * 16 + T - 1) / T;
  const int gTiles = ((NITEMS / 16) + 7) / 8;
  const int gMuT   = ((NUSER / 16) + 7) / 8;

  // pack shared gate weights once
  pack_b16_kernel<<<3, T, 0, stream>>>(wv1W, pw1, 192);
  pack_b16_kernel<<<1, T, 0, stream>>>(wv2W, pw2, 64);
  pack_b16_kernel<<<1, T, 0, stream>>>(wv3W, pw3, 64);

  float* mufA = nullptr; float* sumA = nullptr;
  float* mufB = nullptr; float* sumB = nullptr;

  for (int dom = 0; dom < 2; ++dom) {
    const int*   adj_idx = (const int*)  d_in[dom ? 2 : 0];
    const float* adj_val = (const float*)d_in[dom ? 3 : 1];
    const int*   vp_idx  = (const int*)  d_in[dom ? 8 : 4];
    const float* vp_val  = (const float*)d_in[dom ? 9 : 5];
    const int*   vc_idx  = (const int*)  d_in[dom ? 10 : 6];
    const float* vc_val  = (const float*)d_in[dom ? 11 : 7];
    const float* emb  = (const float*)d_in[dom ? 13 : 12];
    const float* pri  = (const float*)d_in[dom ? 15 : 14];
    const float* cat  = (const float*)d_in[dom ? 17 : 16];
    const float* usr  = (const float*)d_in[dom ? 19 : 18];
    const float* cnv  = (const float*)d_in[dom ? 21 : 20];
    const float* linW = (const float*)d_in[dom ? 24 : 22];  // [2][64][64]
    const float* linb = (const float*)d_in[dom ? 25 : 23];  // [2][64]
    _Float16* plin = dom ? plin1 : plin0;
    _Float16* pusr = dom ? pusr1 : pusr0;

    float* mpb  = ws + dom * DSTR + 0 * SZ;
    float* mcb  = ws + dom * DSTR + 1 * SZ;
    float* e2   = ws + dom * DSTR + 2 * SZ;
    float* item = ws + dom * DSTR + 3 * SZ;
    float* sum  = ws + dom * DSTR + 4 * SZ;
    float* adj  = ws + dom * DSTR + 5 * SZ;
    float* mu2  = ws + dom * DSTR + 6 * SZ;
    float* muf  = mu2 + MU;
    if (dom) { mufB = muf; sumB = sum; } else { mufA = muf; sumA = sum; }

    // pack per-domain B operands: layer-1 linear weight and user matrix
    pack_b16_kernel<<<1, T, 0, stream>>>(linW + 64 * 64, plin, 64);
    pack_b16_kernel<<<128, T, 0, stream>>>(usr, pusr, NUSER);

    // m_p / m_c (layer-invariant gathers)
    gather_scale_kernel<<<gElem, T, 0, stream>>>(vp_idx, vp_val, pri, mpb, NITEMS);
    gather_scale_kernel<<<gElem, T, 0, stream>>>(vc_idx, vc_val, cat, mcb, NITEMS);
    // sum = emb
    copy4_kernel<<<gElem, T, 0, stream>>>((const float4*)emb, (float4*)sum, NITEMS * 16);

    // layer 0: adj spmm on emb, then gate -> item1
    zero4_kernel<<<gElem, T, 0, stream>>>((float4*)adj, NITEMS * 16);
    spmm_atomic_kernel<<<gNNZ, T, 0, stream>>>(adj_idx, adj_val, emb, adj, NNZ_ADJ);
    gate_kernel<<<gTiles, T, 0, stream>>>(emb, mpb, mcb, adj, pw1, wv1b, pw2, wv2b,
                                          pw3, wv3b, item, sum, NITEMS);

    // layer 1: e2 = relu(item1 @ linW[1] + linb[1]) + item1
    lin_res_kernel<<<gTiles, T, 0, stream>>>(item, plin, linb + 64, e2, NITEMS);
    zero4_kernel<<<gElem, T, 0, stream>>>((float4*)adj, NITEMS * 16);
    spmm_atomic_kernel<<<gNNZ, T, 0, stream>>>(adj_idx, adj_val, e2, adj, NNZ_ADJ);
    gate_kernel<<<gTiles, T, 0, stream>>>(e2, mpb, mcb, adj, pw1, wv1b, pw2, wv2b,
                                          pw3, wv3b, item, sum, NITEMS);

    // user path: Mu2 = conv @ user ; MuF = relu(Mu2 @ linW[1] + linb[1]) + Mu2
    conv_gemm_kernel<<<NUSER / 64, 128, 0, stream>>>(cnv, pusr, mu2);
    lin_res_kernel<<<gMuT, T, 0, stream>>>(mu2, plin, linb + 64, muf, NUSER);
  }

  score_kernel<<<(NBATCH + T - 1) / T, T, 0, stream>>>(
      mufA, sumA, (const int*)d_in[32], (const int*)d_in[34],
      mufB, sumB, (const int*)d_in[33], (const int*)d_in[35],
      (float*)d_out);
}
